// TFLayoutLMv3SelfAttention_6296422056245
// MI455X (gfx1250) — matrix operand order repaired and checked by reading, not compiled, not run on hardware
//
#include <hip/hip_runtime.h>

// ---------------------------------------------------------------------------
// TFLayoutLMv3 self-attention, fused flash-attention style, bf16 WMMA (gfx1250)
// B=2, S=1024, HID=768, H=12, D=64, norm=8, cogview softmax == safe softmax
// ---------------------------------------------------------------------------

typedef __bf16 v16bf __attribute__((ext_vector_type(16)));
typedef float  v8f   __attribute__((ext_vector_type(8)));

union Frag { v16bf v; unsigned int u[8]; };

__device__ __forceinline__ unsigned short f2bf(float x) {
    unsigned int u = __float_as_uint(x);
    unsigned int r = (u >> 16) & 1u;
    u += 0x7FFFu + r;                 // round-to-nearest-even
    return (unsigned short)(u >> 16);
}

__device__ __forceinline__ v8f vzero() {
    v8f z = {0.f, 0.f, 0.f, 0.f, 0.f, 0.f, 0.f, 0.f};
    return z;
}

// ---- prep: f32 -> bf16 copy of hidden_states --------------------------------
__global__ void cvt_hidden_kernel(const float* __restrict__ x,
                                  unsigned short* __restrict__ y, int n) {
    int i = blockIdx.x * blockDim.x + threadIdx.x;
    if (i < n) y[i] = f2bf(x[i]);
}

// ---- prep: W[k][n] f32 -> WT[n][k] bf16 ------------------------------------
__global__ void transpose_w_kernel(const float* __restrict__ w,
                                   unsigned short* __restrict__ wt) {
    int n = blockIdx.x * 16 + threadIdx.x;
    int k = blockIdx.y * 16 + threadIdx.y;
    wt[n * 768 + k] = f2bf(w[k * 768 + n]);
}

// ---- QKV projection GEMM: one wave computes a 16(M) x 64(N) tile -----------
// X bf16 [2048][768] row-major, WT bf16 [n][k]. z = 0:Q (scale 1/8), 1:K, 2:V.
// Q,K stored [b][h][s][d] bf16; V stored transposed [b][h][d][s] bf16.
__global__ __launch_bounds__(32) void gemm_qkv_kernel(
    const unsigned short* __restrict__ Xb,
    const unsigned short* __restrict__ WT,
    const float* __restrict__ bq, const float* __restrict__ bk,
    const float* __restrict__ bv,
    unsigned short* __restrict__ Qb, unsigned short* __restrict__ Kb,
    unsigned short* __restrict__ VT)
{
    const int lane = threadIdx.x;
    const int hf   = lane >> 4;   // half-wave id
    const int l16  = lane & 15;
    const int m0   = blockIdx.x * 16;
    const int n0   = blockIdx.y * 64;
    const int z    = blockIdx.z;
    const unsigned short* wt = WT + (size_t)z * 768 * 768;
    const float* bias = (z == 0) ? bq : (z == 1) ? bk : bv;

    v8f acc[4];
    #pragma unroll
    for (int t = 0; t < 4; ++t) acc[t] = vzero();

    const unsigned short* aptr = Xb + (size_t)(m0 + l16) * 768 + hf * 16;

    for (int kb = 0; kb < 768; kb += 32) {
        Frag a;
        #pragma unroll
        for (int i = 0; i < 8; ++i) {
            int off = 2 * i;   // k-offsets {0,2,4,...,14}
            a.u[i] = *(const unsigned int*)(aptr + kb + off);
        }
        #pragma unroll
        for (int t = 0; t < 4; ++t) {
            Frag bfr;
            const unsigned short* bp =
                wt + (size_t)(n0 + 16 * t + l16) * 768 + kb + hf * 16;
            #pragma unroll
            for (int i = 0; i < 8; ++i)
                bfr.u[i] = *(const unsigned int*)(bp + 2 * i);
            acc[t] = __builtin_amdgcn_wmma_f32_16x16x32_bf16(
                false, a.v, false, bfr.v, (short)0, acc[t], false, false);
        }
    }

    const float scale = (z == 0) ? 0.125f : 1.0f;  // q /= sqrt(64)
    #pragma unroll
    for (int t = 0; t < 4; ++t) {
        #pragma unroll
        for (int r = 0; r < 8; ++r) {
            int m  = m0 + r + 8 * hf;
            int nn = n0 + 16 * t + l16;
            float val = (acc[t][r] + bias[nn]) * scale;
            int b = m >> 10, s = m & 1023;
            int h = nn >> 6, d = nn & 63;
            unsigned short o16 = f2bf(val);
            size_t bh = (size_t)b * 12 + h;
            if (z == 0)      Qb[(bh * 1024 + s) * 64 + d] = o16;
            else if (z == 1) Kb[(bh * 1024 + s) * 64 + d] = o16;
            else             VT[(bh * 64 + d) * 1024 + s] = o16;
        }
    }
}

// ---- fused attention: scores + rel bias + online softmax + P@V -------------
// One wave = 16 query rows; 4 waves/block; grid (S/64, B*H).
__global__ __launch_bounds__(128) void attn_kernel(
    const unsigned short* __restrict__ Qb,
    const unsigned short* __restrict__ Kb,
    const unsigned short* __restrict__ VT,
    const float* __restrict__ rel1,
    const float* __restrict__ rel2,
    const float* __restrict__ amask,
    const float* __restrict__ hmask,
    float* __restrict__ out)
{
    __shared__ unsigned short Plds[4][16 * 32];   // per-wave P staging (bf16)

    const int lane = threadIdx.x & 31;
    const int wv   = threadIdx.x >> 5;
    const int hf   = lane >> 4;
    const int l16  = lane & 15;
    const int bh   = blockIdx.y;          // b*12 + h
    const int b    = bh / 12, h = bh % 12;
    const int q0   = blockIdx.x * 64 + wv * 16;

    const unsigned short* qbase = Qb + ((size_t)bh * 1024 + q0 + l16) * 64;
    const unsigned short* kbase = Kb + (size_t)bh * 1024 * 64;
    const unsigned short* vbase = VT + (size_t)bh * 64 * 1024;

    // Q A-fragments for d=0..31 and d=32..63 (loaded once)
    Frag qa[2];
    #pragma unroll
    for (int f = 0; f < 2; ++f) {
        const unsigned short* qp = qbase + 32 * f + hf * 16;
        #pragma unroll
        for (int i = 0; i < 8; ++i) {
            int off = 2 * i;
            qa[f].u[i] = *(const unsigned int*)(qp + off);
        }
    }

    v8f o[4];
    #pragma unroll
    for (int n = 0; n < 4; ++n) o[n] = vzero();
    float mrow[8], lrow[8];
    #pragma unroll
    for (int r = 0; r < 8; ++r) { mrow[r] = -3.0e38f; lrow[r] = 0.f; }

    for (int t0 = 0; t0 < 1024; t0 += 32) {
        // ---- S = Q K^T for two 16-col tiles (K-dim = D = 64 -> 2 chained wmma)
        v8f s[2];
        #pragma unroll
        for (int j = 0; j < 2; ++j) {
            const unsigned short* kp =
                kbase + (size_t)(t0 + 16 * j + l16) * 64 + hf * 16;
            Frag kf0, kf1;
            #pragma unroll
            for (int i = 0; i < 8; ++i) {
                kf0.u[i] = *(const unsigned int*)(kp + 2 * i);
                kf1.u[i] = *(const unsigned int*)(kp + 32 + 2 * i);
            }
            v8f z = vzero();
            z = __builtin_amdgcn_wmma_f32_16x16x32_bf16(
                    false, qa[0].v, false, kf0.v, (short)0, z, false, false);
            z = __builtin_amdgcn_wmma_f32_16x16x32_bf16(
                    false, qa[1].v, false, kf1.v, (short)0, z, false, false);
            s[j] = z;
        }

        // ---- + (rel_pos + rel_2d_pos)/8 + attention_mask  (streamed once, NT)
        float mk0 = amask[b * 1024 + t0 + l16];
        float mk1 = amask[b * 1024 + t0 + 16 + l16];
        #pragma unroll
        for (int r = 0; r < 8; ++r) {
            size_t ri = ((size_t)bh * 1024 + (q0 + r + 8 * hf)) * 1024 + t0 + l16;
            float r0 = (__builtin_nontemporal_load(rel1 + ri) +
                        __builtin_nontemporal_load(rel2 + ri)) * 0.125f;
            float r1 = (__builtin_nontemporal_load(rel1 + ri + 16) +
                        __builtin_nontemporal_load(rel2 + ri + 16)) * 0.125f;
            s[0][r] += r0 + mk0;
            s[1][r] += r1 + mk1;
        }

        // ---- online softmax (per-row stats; 16-lane half-wave reductions)
        #pragma unroll
        for (int r = 0; r < 8; ++r) {
            float mb = fmaxf(s[0][r], s[1][r]);
            #pragma unroll
            for (int d = 1; d < 16; d <<= 1)
                mb = fmaxf(mb, __shfl_xor(mb, d, 32));
            float mn   = fmaxf(mrow[r], mb);
            float corr = __expf(mrow[r] - mn);
            mrow[r] = mn;
            float p0 = __expf(s[0][r] - mn);
            float p1 = __expf(s[1][r] - mn);
            s[0][r] = p0; s[1][r] = p1;
            float rs = p0 + p1;
            #pragma unroll
            for (int d = 1; d < 16; d <<= 1)
                rs += __shfl_xor(rs, d, 32);
            lrow[r] = lrow[r] * corr + rs;
            #pragma unroll
            for (int n = 0; n < 4; ++n) o[n][r] *= corr;
        }

        // ---- re-shape P: C-layout -> LDS (16x32 bf16, row-major) -> A-layout
        #pragma unroll
        for (int r = 0; r < 8; ++r) {
            int row = r + 8 * hf;
            Plds[wv][row * 32 + l16]      = f2bf(s[0][r]);
            Plds[wv][row * 32 + 16 + l16] = f2bf(s[1][r]);
        }
        Frag pa;
        {
            const unsigned short* pp = &Plds[wv][l16 * 32 + hf * 16];
            #pragma unroll
            for (int i = 0; i < 8; ++i) {
                int off = 2 * i;
                pa.u[i] = *(const unsigned int*)(pp + off);
            }
        }

        // ---- O += P @ V   (V transposed [d][s] -> contiguous B-fragments)
        #pragma unroll
        for (int n = 0; n < 4; ++n) {
            const unsigned short* vp =
                vbase + (size_t)(16 * n + l16) * 1024 + t0 + hf * 16;
            Frag vf;
            #pragma unroll
            for (int i = 0; i < 8; ++i)
                vf.u[i] = *(const unsigned int*)(vp + 2 * i);
            o[n] = __builtin_amdgcn_wmma_f32_16x16x32_bf16(
                false, pa.v, false, vf.v, (short)0, o[n], false, false);
        }
    }

    // ---- normalize, apply head_mask, write ctx [B,S,H*D] f32
    float hm = hmask[h];
    #pragma unroll
    for (int r = 0; r < 8; ++r) {
        float inv = hm / lrow[r];
        int srow  = q0 + r + 8 * hf;
        #pragma unroll
        for (int n = 0; n < 4; ++n) {
            out[((size_t)b * 1024 + srow) * 768 + h * 64 + 16 * n + l16] =
                o[n][r] * inv;
        }
    }
}

// ---------------------------------------------------------------------------
extern "C" void kernel_launch(void* const* d_in, const int* in_sizes, int n_in,
                              void* d_out, int out_size, void* d_ws, size_t ws_size,
                              hipStream_t stream) {
    const float* hidden = (const float*)d_in[0];
    const float* amask  = (const float*)d_in[1];
    const float* hmask  = (const float*)d_in[2];
    const float* rel1   = (const float*)d_in[3];
    const float* rel2   = (const float*)d_in[4];
    const float* Wq = (const float*)d_in[5];
    const float* bq = (const float*)d_in[6];
    const float* Wk = (const float*)d_in[7];
    const float* bk = (const float*)d_in[8];
    const float* Wv = (const float*)d_in[9];
    const float* bv = (const float*)d_in[10];
    float* out = (float*)d_out;

    char*  ws  = (char*)d_ws;
    size_t off = 0;
    auto carve = [&](size_t bytes) -> char* {
        char* p = ws + off;
        off += (bytes + 255) & ~(size_t)255;
        return p;
    };
    unsigned short* Xb = (unsigned short*)carve(2048ull * 768 * 2);      // hidden bf16
    unsigned short* WT = (unsigned short*)carve(3ull * 768 * 768 * 2);   // W^T bf16 x3
    unsigned short* Qb = (unsigned short*)carve(24ull * 1024 * 64 * 2);  // Q/8 bf16
    unsigned short* Kb = (unsigned short*)carve(24ull * 1024 * 64 * 2);  // K bf16
    unsigned short* VT = (unsigned short*)carve(24ull * 1024 * 64 * 2);  // V^T bf16

    cvt_hidden_kernel<<<(2048 * 768 + 255) / 256, 256, 0, stream>>>(
        hidden, Xb, 2048 * 768);

    dim3 tg(768 / 16, 768 / 16);
    transpose_w_kernel<<<tg, dim3(16, 16), 0, stream>>>(Wq, WT + 0ull * 768 * 768);
    transpose_w_kernel<<<tg, dim3(16, 16), 0, stream>>>(Wk, WT + 1ull * 768 * 768);
    transpose_w_kernel<<<tg, dim3(16, 16), 0, stream>>>(Wv, WT + 2ull * 768 * 768);

    gemm_qkv_kernel<<<dim3(128, 12, 3), 32, 0, stream>>>(
        Xb, WT, bq, bk, bv, Qb, Kb, VT);

    attn_kernel<<<dim3(16, 24), 128, 0, stream>>>(
        Qb, Kb, VT, rel1, rel2, amask, hmask, out);
}
